// Attention_3195455668613
// MI455X (gfx1250) — matrix operand order repaired and checked by reading, not compiled, not run on hardware
//
#include <hip/hip_runtime.h>
#include <hip/hip_bf16.h>

typedef __attribute__((ext_vector_type(16))) __bf16 v16bf;
typedef __attribute__((ext_vector_type(8)))  __bf16 v8bf;
typedef __attribute__((ext_vector_type(8)))  float  v8f;
typedef __attribute__((ext_vector_type(4)))  float  f4;
typedef int v4i __attribute__((vector_size(16)));

constexpr int Bn = 8;
constexpr int Sn = 2048;
constexpr int Dn = 1024;
constexpr int Kn = 64;
constexpr int BS = Bn * Sn;

// ---------------- WMMA helpers (CDNA5 gfx1250, wave32) ----------------

__device__ inline v8f wmma_bf16(v16bf a, v16bf b, v8f c) {
  // v_wmma_f32_16x16x32_bf16: D = A(16x32) * B(32x16) + C(16x16 f32)
  return __builtin_amdgcn_wmma_f32_16x16x32_bf16(false, a, false, b,
                                                 (short)0, c, false, false);
}

// ---- async global -> LDS copy (gfx1250 ASYNCcnt path), with safe fallback ----
__device__ inline void copy16_async(const __bf16* g, __bf16* l) {
#if __has_builtin(__builtin_amdgcn_global_load_async_to_lds_b128)
  __builtin_amdgcn_global_load_async_to_lds_b128(
      (__attribute__((address_space(1))) v4i*)g,
      (__attribute__((address_space(3))) v4i*)l, 0, 0);
#else
  *(uint4*)l = *(const uint4*)g;
#endif
}
__device__ inline void wait_async_copies() {
#if __has_builtin(__builtin_amdgcn_global_load_async_to_lds_b128)
#if __has_builtin(__builtin_amdgcn_s_wait_asynccnt)
  __builtin_amdgcn_s_wait_asynccnt(0);
#else
  asm volatile("s_wait_asynccnt 0x0" ::: "memory");
#endif
#endif
}

// A fragment 16x32 (MxK) bf16 from row-major f32 src, row stride ld.
// Lane L: M = L&15; lanes<16 hold K=0..15 ; lanes>=16 hold K=16..31.
__device__ inline v16bf load_a_f32v(const float* src, int ld, int lane) {
  int m  = lane & 15;
  int kb = (lane >> 4) << 4;          // 0 or 16
  const float* r = src + m * ld + kb;
  f4 x0 = *(const f4*)(r);
  f4 x1 = *(const f4*)(r + 4);
  f4 y0 = *(const f4*)(r + 8);
  f4 y1 = *(const f4*)(r + 12);
  v16bf a;
#pragma unroll
  for (int e = 0; e < 4; ++e) {
    a[e]      = (__bf16)x0[e];
    a[e + 4]  = (__bf16)x1[e];
    a[e + 8]  = (__bf16)y0[e];
    a[e + 12] = (__bf16)y1[e];
  }
  return a;
}

// Same A layout from bf16 source: 2x 16-byte runs.
__device__ inline v16bf load_a_bf16v(const __bf16* src, int ld, int lane) {
  int m  = lane & 15;
  int kb = (lane >> 4) << 4;
  const __bf16* r = src + m * ld + kb;
  v8bf lo = *(const v8bf*)(r);
  v8bf hi = *(const v8bf*)(r + 8);
  v16bf a;
#pragma unroll
  for (int e = 0; e < 8; ++e) { a[e] = lo[e]; a[e + 8] = hi[e]; }
  return a;
}

// B fragment 32x16 (KxN) where memory holds the TRANSPOSE:
// element (k,n) = src[n*ld + k].  One contiguous 32-byte read per lane.
__device__ inline v16bf load_bTv(const __bf16* src, int ld, int lane) {
  int n  = lane & 15;
  int kb = (lane >> 4) << 4;          // 0 or 16
  return *(const v16bf*)(src + n * ld + kb);
}

// B fragment from fragment-major repacked weights: [frag][lane][16] contiguous.
__device__ inline v16bf load_frag(const __bf16* base, int lane) {
  return *(const v16bf*)(base + lane * 16);
}

// ---------------- Kernel 0: repack weights to bf16, fragment-major ----------------
// Wq/Wk/Wv go to ONE combined chunk-contiguous stream:
//   wqkvf[((c*12 + w*4 + nt)*32 + lane)*16 + e], 32 chunks x 12 frags = 12KB per chunk.
// Wo: separate, [chunk(2)][ntile(64)][lane][16].
__global__ void repack_kernel(const float* __restrict__ Wq, const float* __restrict__ Wk,
                              const float* __restrict__ Wv, const float* __restrict__ Wo,
                              __bf16* __restrict__ wqkvf, __bf16* __restrict__ wof) {
  int t = blockIdx.x * blockDim.x + threadIdx.x;    // one fragment (16 elems) per thread
  int which = t >> 12;                              // 0..3
  int rem   = t & 4095;
  int lane  = rem & 31;
  int fr    = rem >> 5;
  const float* srcs[4] = {Wq, Wk, Wv, Wo};
  int ldw, c, nt;
  __bf16* dst;
  if (which == 3) {
    ldw = Dn; nt = fr & 63; c = fr >> 6;
    dst = wof + rem * 16;
  } else {
    ldw = Kn; nt = fr & 3;  c = fr >> 2;
    dst = wqkvf + ((c * 12 + which * 4 + nt) * 32 + lane) * 16;
  }
  int n  = lane & 15;
  int kb = (lane >> 4) << 4;
  const float* src = srcs[which];
  v16bf f;
#pragma unroll
  for (int e = 0; e < 16; ++e)
    f[e] = (__bf16)src[(c * 32 + kb + e) * ldw + nt * 16 + n];
  *(v16bf*)dst = f;
}

// ---------------- Kernel 1: QKV projection, LDS-staged weights ----------------
// grid = BS/64 blocks x 128 threads (4 waves); each wave owns a 16-row tile.
// Weights double-buffered in LDS via async copies; all 4 waves share them.
// q,k stored row-major bf16 [BS,64]; v stored TRANSPOSED per batch: vT[(b*64+col)*Sn + s].
__global__ void qkv_kernel(const float* __restrict__ X,
                           const __bf16* __restrict__ wqkvf,
                           const float* __restrict__ bq, const float* __restrict__ bk,
                           const float* __restrict__ bv,
                           __bf16* __restrict__ qb, __bf16* __restrict__ kbuf,
                           __bf16* __restrict__ vT) {
  __shared__ __bf16 wbuf[2][12 * 512];      // 2 x 12KB weight staging
  int lane = threadIdx.x & 31;
  int wave = threadIdx.x >> 5;
  int t = threadIdx.x;
  int rowBase = blockIdx.x * 64 + wave * 16;

  // prologue: stage chunk 0 into buffer 0 (768 x 16B segments, 6 per thread)
#pragma unroll
  for (int i = 0; i < 6; ++i) {
    int seg = t + i * 128;
    copy16_async(wqkvf + seg * 8, &wbuf[0][seg * 8]);
  }
  wait_async_copies();
  __syncthreads();

  v8f acc[3][4];
#pragma unroll
  for (int w = 0; w < 3; ++w)
#pragma unroll
    for (int nt = 0; nt < 4; ++nt) acc[w][nt] = (v8f){0,0,0,0,0,0,0,0};

  v16bf a = load_a_f32v(X + rowBase * Dn, Dn, lane);

  for (int c = 0; c < Dn / 32; ++c) {
    int cur = c & 1;
    // kick off async staging of the NEXT chunk while we compute this one
    if (c < Dn / 32 - 1) {
      const __bf16* src = wqkvf + (c + 1) * 6144;
#pragma unroll
      for (int i = 0; i < 6; ++i) {
        int seg = t + i * 128;
        copy16_async(src + seg * 8, &wbuf[cur ^ 1][seg * 8]);
      }
    }
    // prefetch next A fragment across the WMMA burst
    v16bf an = a;
    if (c < Dn / 32 - 1)
      an = load_a_f32v(X + rowBase * Dn + (c + 1) * 32, Dn, lane);

#pragma unroll
    for (int w = 0; w < 3; ++w)
#pragma unroll
      for (int nt = 0; nt < 4; ++nt) {
        v16bf b = load_frag(&wbuf[cur][(w * 4 + nt) * 512], lane);
        acc[w][nt] = wmma_bf16(a, b, acc[w][nt]);
      }

    if (c < Dn / 32 - 1) {
      wait_async_copies();
      __syncthreads();
    }
    a = an;
  }

  int n = lane & 15, hi = lane >> 4;
  int batch = rowBase / Sn;
  int rowIn = rowBase % Sn;

  // q, k: row-major stores
  const float* biases[2] = {bq, bk};
  __bf16* outs[2] = {qb, kbuf};
#pragma unroll
  for (int w = 0; w < 2; ++w) {
#pragma unroll
    for (int nt = 0; nt < 4; ++nt) {
      int col = nt * 16 + n;
      float bias = biases[w][col];
#pragma unroll
      for (int i = 0; i < 8; ++i) {
        int row = i + 8 * hi;
        outs[w][(rowBase + row) * Kn + col] = (__bf16)(acc[w][nt][i] + bias);
      }
    }
  }
  // v: transposed, packed 16B store per lane (rows contiguous along s)
#pragma unroll
  for (int nt = 0; nt < 4; ++nt) {
    int col = nt * 16 + n;
    float bias = bv[col];
    v8bf pack;
#pragma unroll
    for (int i = 0; i < 8; ++i) pack[i] = (__bf16)(acc[2][nt][i] + bias);
    *(v8bf*)(vT + ((size_t)batch * Kn + col) * Sn + rowIn + 8 * hi) = pack;
  }
}

// ---------------- Kernel 2: attention (scores -> softmax -> P@V) ----------------
// grid = Bn*(Sn/16) blocks x 256 threads (8 waves). LDS: 16 x Sn f32 = 128 KB.
__global__ void attn_kernel(const __bf16* __restrict__ qb,
                            const __bf16* __restrict__ kbuf,
                            const __bf16* __restrict__ vT,
                            __bf16* __restrict__ avb) {
  extern __shared__ float sc[];      // [16][Sn] scores / exp / partials
  __shared__ float ssum[16];

  int lane = threadIdx.x & 31;
  int wave = threadIdx.x >> 5;
  int qt    = blockIdx.x % (Sn / 16);
  int batch = blockIdx.x / (Sn / 16);
  int qRow  = batch * Sn + qt * 16;

  int n = lane & 15, hi = lane >> 4;

  // q tile A-fragments (head dim 64 = two K=32 chunks)
  v16bf aq0 = load_a_bf16v(qb + qRow * Kn + 0,  Kn, lane);
  v16bf aq1 = load_a_bf16v(qb + qRow * Kn + 32, Kn, lane);

  // --- scores: each wave covers 256 keys; 2 key-tiles (32 keys) per iteration ---
  for (int kt = 0; kt < 16; kt += 2) {
    int key0 = wave * 256 + kt * 16;
    int key1 = key0 + 16;
    const __bf16* kp0 = kbuf + (batch * Sn + key0) * Kn;
    const __bf16* kp1 = kbuf + (batch * Sn + key1) * Kn;
    v16bf b00 = load_bTv(kp0 + 0,  Kn, lane);
    v16bf b01 = load_bTv(kp0 + 32, Kn, lane);
    v16bf b10 = load_bTv(kp1 + 0,  Kn, lane);
    v16bf b11 = load_bTv(kp1 + 32, Kn, lane);
    v8f acc0 = (v8f){0,0,0,0,0,0,0,0};
    v8f acc1 = (v8f){0,0,0,0,0,0,0,0};
    acc0 = wmma_bf16(aq0, b00, acc0);
    acc1 = wmma_bf16(aq0, b10, acc1);
    acc0 = wmma_bf16(aq1, b01, acc0);
    acc1 = wmma_bf16(aq1, b11, acc1);
#pragma unroll
    for (int i = 0; i < 8; ++i) {
      sc[(i + 8 * hi) * Sn + key0 + n] = acc0[i] * 0.125f;   // 1/sqrt(64)
      sc[(i + 8 * hi) * Sn + key1 + n] = acc1[i] * 0.125f;
    }
  }
  __syncthreads();

  // --- softmax stats: 16 lanes per row, exp stored unnormalized in place ---
  {
    int row = threadIdx.x >> 4;
    int l16 = threadIdx.x & 15;
    float m = -1e30f;
    for (int j = 0; j < Sn / 16; ++j)
      m = fmaxf(m, sc[row * Sn + l16 + j * 16]);
#pragma unroll
    for (int o = 8; o >= 1; o >>= 1) m = fmaxf(m, __shfl_xor(m, o, 32));
    float s = 0.f;
    for (int j = 0; j < Sn / 16; ++j) {
      float e = __expf(sc[row * Sn + l16 + j * 16] - m);
      sc[row * Sn + l16 + j * 16] = e;
      s += e;
    }
#pragma unroll
    for (int o = 8; o >= 1; o >>= 1) s += __shfl_xor(s, o, 32);
    if (l16 == 0) ssum[row] = s;
  }
  __syncthreads();

  // --- P@V: each wave accumulates over its 256-key range; 64 keys per iteration ---
  v8f accv[4];
#pragma unroll
  for (int nt = 0; nt < 4; ++nt) accv[nt] = (v8f){0,0,0,0,0,0,0,0};

  const __bf16* vbase = vT + (size_t)batch * Kn * Sn;
  for (int j = 0; j < 8; j += 2) {
    int s0 = wave * 256 + j * 32;
    int s1 = s0 + 32;
    v16bf ap0 = load_a_f32v(sc + s0, Sn, lane);   // probs (unnormalized) from LDS
    v16bf ap1 = load_a_f32v(sc + s1, Sn, lane);
    v16bf vb[8];
#pragma unroll
    for (int nt = 0; nt < 4; ++nt) {
      vb[nt]     = load_bTv(vbase + (size_t)nt * 16 * Sn + s0, Sn, lane);
      vb[4 + nt] = load_bTv(vbase + (size_t)nt * 16 * Sn + s1, Sn, lane);
    }
#pragma unroll
    for (int nt = 0; nt < 4; ++nt) accv[nt] = wmma_bf16(ap0, vb[nt], accv[nt]);
#pragma unroll
    for (int nt = 0; nt < 4; ++nt) accv[nt] = wmma_bf16(ap1, vb[4 + nt], accv[nt]);
  }
  __syncthreads();                    // done reading score LDS

  // cross-wave reduction of the 8 partial 16x64 tiles (reuse sc)
  float* part = sc;
#pragma unroll
  for (int nt = 0; nt < 4; ++nt)
#pragma unroll
    for (int i = 0; i < 8; ++i)
      part[wave * 1024 + (i + 8 * hi) * 64 + nt * 16 + n] = accv[nt][i];
  __syncthreads();

#pragma unroll
  for (int j = 0; j < 4; ++j) {
    int idx = threadIdx.x + j * 256;  // 0..1023 over 16x64
    float s = 0.f;
#pragma unroll
    for (int w = 0; w < 8; ++w) s += part[w * 1024 + idx];
    int row = idx >> 6;
    avb[(qRow + row) * Kn + (idx & 63)] = (__bf16)(s / ssum[row]);
  }
}

// ---------------- Kernel 3: out projection + residual + LayerNorm ----------------
// grid = BS/16 blocks x 256 threads (8 waves). LDS: 16 x Dn f32 = 64 KB.
__global__ void proj_ln_kernel(const float* __restrict__ X,
                               const __bf16* __restrict__ avb,
                               const __bf16* __restrict__ wof,
                               const float* __restrict__ bo,
                               const float* __restrict__ gamma,
                               const float* __restrict__ beta,
                               float* __restrict__ out) {
  extern __shared__ float yt[];      // [16][Dn]
  int lane = threadIdx.x & 31;
  int wave = threadIdx.x >> 5;
  int rowBase = blockIdx.x * 16;
  int n = lane & 15, hi = lane >> 4;

  v16bf a0 = load_a_bf16v(avb + rowBase * Kn + 0,  Kn, lane);
  v16bf a1 = load_a_bf16v(avb + rowBase * Kn + 32, Kn, lane);

  v8f acc[8];
#pragma unroll
  for (int nt = 0; nt < 8; ++nt) acc[nt] = (v8f){0,0,0,0,0,0,0,0};
#pragma unroll
  for (int nt = 0; nt < 8; ++nt) {
    int ntg = wave * 8 + nt;                 // global ntile 0..63
    acc[nt] = wmma_bf16(a0, load_frag(wof + (0 * 64 + ntg) * 512, lane), acc[nt]);
    acc[nt] = wmma_bf16(a1, load_frag(wof + (1 * 64 + ntg) * 512, lane), acc[nt]);
  }

  // y = X + av@Wo + bo  -> LDS
#pragma unroll
  for (int nt = 0; nt < 8; ++nt) {
    int col = wave * 128 + nt * 16 + n;
    float bias = bo[col];
#pragma unroll
    for (int i = 0; i < 8; ++i) {
      int row = i + 8 * hi;
      yt[row * Dn + col] = acc[nt][i] + bias + X[(rowBase + row) * Dn + col];
    }
  }
  __syncthreads();

  // LayerNorm: 16 lanes per row
  int row = threadIdx.x >> 4;
  int l16 = threadIdx.x & 15;
  float s = 0.f, s2 = 0.f;
  for (int j = 0; j < Dn / 16; ++j) {
    float v = yt[row * Dn + l16 + j * 16];
    s += v; s2 += v * v;
  }
#pragma unroll
  for (int o = 8; o >= 1; o >>= 1) {
    s  += __shfl_xor(s,  o, 32);
    s2 += __shfl_xor(s2, o, 32);
  }
  float mu   = s * (1.f / Dn);
  float var  = s2 * (1.f / Dn) - mu * mu;
  float rstd = rsqrtf(var + 1e-5f);
  for (int j = 0; j < Dn / 16; ++j) {
    int c = l16 + j * 16;
    float v = yt[row * Dn + c];
    out[(rowBase + row) * Dn + c] = (v - mu) * rstd * gamma[c] + beta[c];
  }
}

// ---------------- launch ----------------
extern "C" void kernel_launch(void* const* d_in, const int* in_sizes, int n_in,
                              void* d_out, int out_size, void* d_ws, size_t ws_size,
                              hipStream_t stream) {
  const float* X     = (const float*)d_in[0];
  const float* Wq    = (const float*)d_in[1];
  const float* bq    = (const float*)d_in[2];
  const float* Wk    = (const float*)d_in[3];
  const float* bk    = (const float*)d_in[4];
  const float* Wv    = (const float*)d_in[5];
  const float* bv    = (const float*)d_in[6];
  const float* Wo    = (const float*)d_in[7];
  const float* bo    = (const float*)d_in[8];
  const float* gamma = (const float*)d_in[9];
  const float* beta  = (const float*)d_in[10];
  float* out = (float*)d_out;

  __bf16* qb    = (__bf16*)d_ws;                       // BS*64 bf16
  __bf16* kbuf  = qb    + (size_t)BS * Kn;             // BS*64
  __bf16* vT    = kbuf  + (size_t)BS * Kn;             // BS*64 (transposed per batch)
  __bf16* avb   = vT    + (size_t)BS * Kn;             // BS*64
  __bf16* wqkvf = avb   + (size_t)BS * Kn;             // 3*65536 combined stream
  __bf16* wof   = wqkvf + 3 * 65536;                   // 65536

  repack_kernel<<<64, 256, 0, stream>>>(Wq, Wk, Wv, Wo, wqkvf, wof);
  qkv_kernel<<<BS / 64, 128, 0, stream>>>(X, wqkvf, bq, bk, bv, qb, kbuf, vT);
  attn_kernel<<<Bn * (Sn / 16), 256, 16 * Sn * sizeof(float), stream>>>(qb, kbuf, vT, avb);
  proj_ln_kernel<<<BS / 16, 256, 16 * Dn * sizeof(float), stream>>>(X, avb, wof, bo, gamma,
                                                                    beta, out);
}
